// OuterProductMean_21741124452584
// MI455X (gfx1250) — compile-verified, matrix operand order checked by reading, not run on hardware
//
#include <hip/hip_runtime.h>

#define S_DIM 128
#define N_DIM 384
#define CM    256
#define CH    32
#define CZ    128

typedef __attribute__((ext_vector_type(16))) __bf16 v16bf;
typedef __attribute__((ext_vector_type(8)))  float  v8f;

union BF16x16 { uint4 u[2]; v16bf v; };

__device__ __forceinline__ v16bf ld_bf16x16(const void* p) {
  BF16x16 t;
  t.u[0] = *(const uint4*)p;
  t.u[1] = *(const uint4*)((const char*)p + 16);
  return t.v;
}

// CDNA5 async copy: 16B global -> LDS, tracked by ASYNCcnt (no VGPR data path).
// lds_off = wave-relative LDS byte address (low 32 bits of flat LDS pointer).
__device__ __forceinline__ void async_g2lds_b128(unsigned lds_off, const void* gptr) {
  asm volatile("global_load_async_to_lds_b128 %0, %1, off"
               :: "v"(lds_off), "v"(gptr) : "memory");
}
__device__ __forceinline__ void wait_asynccnt0() {
  asm volatile("s_wait_asynccnt 0" ::: "memory");
}

// ---------------------------------------------------------------------------
// Kernel 0: build bf16 transposed weights.
//   WcatT[c][k] (64x256)  = c<32 ? Wl[k,c] : Wr[k,c-32]
//   WoT[z][k]  (128x1024) = Wo[k,z]
// ---------------------------------------------------------------------------
__global__ void prep_kernel(const float* __restrict__ Wl,
                            const float* __restrict__ Wr,
                            const float* __restrict__ Wo,
                            __bf16* __restrict__ WcatT,
                            __bf16* __restrict__ WoT) {
  int idx = blockIdx.x * blockDim.x + threadIdx.x;
  int stride = gridDim.x * blockDim.x;
  for (int i = idx; i < 64 * 256; i += stride) {
    int c = i >> 8, k = i & 255;
    float v = (c < 32) ? Wl[k * 32 + c] : Wr[k * 32 + (c - 32)];
    WcatT[i] = (__bf16)v;
  }
  for (int i = idx; i < 128 * 1024; i += stride) {
    int z = i >> 10, k = i & 1023;
    WoT[i] = (__bf16)Wo[k * 128 + z];
  }
}

// ---------------------------------------------------------------------------
// Kernel 1: fused LayerNorm + left/right projection (WMMA), K-major outputs
//   LT[(n*32+c)*128 + s] = left[s,n,c]   (bf16)
//   RT[(n*32+d)*128 + s] = right[s,n,d]  (bf16)
// 256 threads = 8 waves; 128 rows (s,n) per block; grid = 49152/128 = 384.
// ---------------------------------------------------------------------------
#define K1_STRIDE 528   // 256 bf16 + 16B pad -> conflict-free WMMA-A fetches

__global__ __launch_bounds__(256) void ln_proj_kernel(
    const float* __restrict__ msa, const float* __restrict__ g,
    const float* __restrict__ b, const __bf16* __restrict__ WcatT,
    const float* __restrict__ bl, const float* __restrict__ br,
    __bf16* __restrict__ LT, __bf16* __restrict__ RT) {
  extern __shared__ char smem[];
  char* Abf = smem;                       // 128 rows x K1_STRIDE bytes
  const int tid = threadIdx.x;
  const int w   = tid >> 5;
  const int l   = tid & 31;
  const int l16 = l & 15;
  const int hi  = l >> 4;
  const int rowBase = blockIdx.x * 128;

  float gu[8], bu[8];
#pragma unroll
  for (int u = 0; u < 8; ++u) { gu[u] = g[u * 32 + l]; bu[u] = b[u * 32 + l]; }

  // ---- LayerNorm: wave w owns local rows [w*16, w*16+16) ----
  for (int t = 0; t < 16; ++t) {
    int lrow = w * 16 + t;
    const float* x = msa + (size_t)(rowBase + lrow) * CM;
    float xs[8], sum = 0.f, sq = 0.f;
#pragma unroll
    for (int u = 0; u < 8; ++u) {
      float v = x[u * 32 + l];
      xs[u] = v; sum += v; sq += v * v;
    }
#pragma unroll
    for (int m = 16; m >= 1; m >>= 1) {
      sum += __shfl_xor(sum, m, 32);
      sq  += __shfl_xor(sq,  m, 32);
    }
    float mu   = sum * (1.f / CM);
    float var  = sq * (1.f / CM) - mu * mu;
    float rstd = rsqrtf(var + 1e-5f);
    __bf16* dst = (__bf16*)(Abf + lrow * K1_STRIDE);
#pragma unroll
    for (int u = 0; u < 8; ++u)
      dst[u * 32 + l] = (__bf16)((xs[u] - mu) * rstd * gu[u] + bu[u]);
  }
  __syncthreads();

  // ---- WMMA projection: wave w = M-tile w; K = 256 (8 steps); N = 64 ----
  v16bf Areg[8];
#pragma unroll
  for (int ks = 0; ks < 8; ++ks)
    Areg[ks] = ld_bf16x16(Abf + (w * 16 + l16) * K1_STRIDE + (ks * 32 + hi * 16) * 2);

#pragma unroll 1
  for (int nt = 0; nt < 4; ++nt) {
    v8f acc = {};
    const __bf16* brow = WcatT + (nt * 16 + l16) * 256;
#pragma unroll
    for (int ks = 0; ks < 8; ++ks) {
      v16bf bmat = ld_bf16x16(brow + ks * 32 + hi * 16);
      acc = __builtin_amdgcn_wmma_f32_16x16x32_bf16(false, Areg[ks], false, bmat,
                                                    (short)0, acc, false, false);
    }
    int col  = nt * 16 + l16;
    float bias = (col < 32) ? bl[col] : br[col - 32];
    __bf16* dstbase = (nt < 2) ? LT : RT;
    int c32 = col & 31;
#pragma unroll
    for (int r = 0; r < 8; ++r) {
      int R = rowBase + w * 16 + r + 8 * hi;   // global (s,n) row id = s*384+n
      int s = R / N_DIM;
      int n = R - s * N_DIM;
      dstbase[((size_t)(n * CH + c32)) * S_DIM + s] = (__bf16)(acc[r] + bias);
    }
  }
}

// ---------------------------------------------------------------------------
// Kernel 2: fused outer-product-mean + Wo projection.
// One 8x8 (i,j) block per WG (grid 48x48), 256 threads = 8 waves.
//   Stage:   LT/RT blocks -> LDS via GLOBAL_LOAD_ASYNC_TO_LDS_B128 (ASYNCcnt)
//   Phase A: outer[(i,c),(j,d)] = (1/S) * LTblk @ RTblk^T  (256x256x128 GEMM)
//            -> bf16 tile in LDS, laid out as phase-B A operand rows
//               ot[row=(i_loc*8+j_loc)][k=c*32+d]
//   Phase B: out[64 pairs, 128] = ot @ WoT^T + bo           (64x128x1024 GEMM)
// ---------------------------------------------------------------------------
#define LR_STRIDE 272    // 128 bf16 + 16B pad
#define OT_STRIDE 2064   // 1024 bf16 + 16B pad

__global__ __launch_bounds__(256) void outer_proj_kernel(
    const __bf16* __restrict__ LT, const __bf16* __restrict__ RT,
    const __bf16* __restrict__ WoT, const float* __restrict__ bo,
    float* __restrict__ out) {
  extern __shared__ char smem[];
  char* lt_s = smem;                         // 256 x LR_STRIDE
  char* rt_s = smem + 256 * LR_STRIDE;       // 256 x LR_STRIDE
  char* ot_s = smem + 2 * 256 * LR_STRIDE;   // 64  x OT_STRIDE
  const int tid = threadIdx.x;
  const int w   = tid >> 5;
  const int l   = tid & 31;
  const int l16 = l & 15;
  const int hi  = l >> 4;
  const int i0  = blockIdx.y * 8;
  const int j0  = blockIdx.x * 8;

  // ---- async-stage LT/RT blocks into LDS (256 rows x 256B, padded rows) ----
  {
    const unsigned lt_base = (unsigned)(size_t)lt_s;   // wave-relative LDS addr
    const unsigned rt_base = (unsigned)(size_t)rt_s;
    const char* gl = (const char*)(LT + (size_t)i0 * 32 * S_DIM);
    const char* gr = (const char*)(RT + (size_t)j0 * 32 * S_DIM);
#pragma unroll
    for (int it = 0; it < 16; ++it) {
      int idx = it * 256 + tid;              // 4096 16B segments per array
      int row = idx >> 4, seg = idx & 15;
      unsigned loff = row * LR_STRIDE + seg * 16;
      size_t   goff = (size_t)row * 256 + seg * 16;
      async_g2lds_b128(lt_base + loff, gl + goff);
      async_g2lds_b128(rt_base + loff, gr + goff);
    }
    wait_asynccnt0();
  }
  __syncthreads();

  // ---- Phase A: wave w owns M-tiles {2w, 2w+1} x all 16 N-tiles ----
  const float inv_s = 1.0f / S_DIM;
#pragma unroll 1
  for (int mi = 0; mi < 2; ++mi) {
    int mt = w * 2 + mi;
    v16bf Areg[4];
#pragma unroll
    for (int ks = 0; ks < 4; ++ks)
      Areg[ks] = ld_bf16x16(lt_s + (mt * 16 + l16) * LR_STRIDE + (ks * 32 + hi * 16) * 2);
#pragma unroll 1
    for (int nt = 0; nt < 16; ++nt) {
      v8f acc = {};
#pragma unroll
      for (int ks = 0; ks < 4; ++ks) {
        v16bf bmat = ld_bf16x16(rt_s + (nt * 16 + l16) * LR_STRIDE + (ks * 32 + hi * 16) * 2);
        acc = __builtin_amdgcn_wmma_f32_16x16x32_bf16(false, Areg[ks], false, bmat,
                                                      (short)0, acc, false, false);
      }
      int q = nt * 16 + l16;              // (j,d) index 0..255
      int j_loc = q >> 5, dd = q & 31;
#pragma unroll
      for (int r = 0; r < 8; ++r) {
        int p = mt * 16 + r + 8 * hi;     // (i,c) index 0..255
        int i_loc = p >> 5, cc = p & 31;
        *((__bf16*)(ot_s + (i_loc * 8 + j_loc) * OT_STRIDE) + (cc * 32 + dd)) =
            (__bf16)(acc[r] * inv_s);
      }
    }
  }
  __syncthreads();

  // ---- Phase B: wave w -> M-tile w/2, N-tiles [4*(w&1), 4*(w&1)+4) ----
  int Mt = w >> 1;
  int NtBase = (w & 1) * 4;
  v8f acc[4] = {{}, {}, {}, {}};
#pragma unroll 1
  for (int ks = 0; ks < 32; ++ks) {
    v16bf a = ld_bf16x16(ot_s + (Mt * 16 + l16) * OT_STRIDE + (ks * 32 + hi * 16) * 2);
#pragma unroll
    for (int q = 0; q < 4; ++q) {
      int z = (NtBase + q) * 16 + l16;
      v16bf bmat = ld_bf16x16(WoT + (size_t)z * 1024 + ks * 32 + hi * 16);
      acc[q] = __builtin_amdgcn_wmma_f32_16x16x32_bf16(false, a, false, bmat,
                                                       (short)0, acc[q], false, false);
    }
  }
#pragma unroll
  for (int q = 0; q < 4; ++q) {
    int z = (NtBase + q) * 16 + l16;
    float bias = bo[z];
#pragma unroll
    for (int r = 0; r < 8; ++r) {
      int rowp  = Mt * 16 + r + 8 * hi;   // pair row = i_loc*8 + j_loc
      int i_loc = rowp >> 3, j_loc = rowp & 7;
      out[(((size_t)(i0 + i_loc)) * N_DIM + (j0 + j_loc)) * CZ + z] = acc[q][r] + bias;
    }
  }
}

// ---------------------------------------------------------------------------
extern "C" void kernel_launch(void* const* d_in, const int* in_sizes, int n_in,
                              void* d_out, int out_size, void* d_ws, size_t ws_size,
                              hipStream_t stream) {
  (void)in_sizes; (void)n_in; (void)out_size; (void)ws_size;
  const float* msa  = (const float*)d_in[0];
  const float* ln_g = (const float*)d_in[1];
  const float* ln_b = (const float*)d_in[2];
  const float* Wl   = (const float*)d_in[3];
  const float* bl   = (const float*)d_in[4];
  const float* Wr   = (const float*)d_in[5];
  const float* br   = (const float*)d_in[6];
  const float* Wo   = (const float*)d_in[7];
  const float* bo   = (const float*)d_in[8];
  float* out = (float*)d_out;

  char* ws = (char*)d_ws;
  const size_t lt_bytes = (size_t)N_DIM * CH * S_DIM * 2;   // 3 MB
  __bf16* LT    = (__bf16*)ws;
  __bf16* RT    = (__bf16*)(ws + lt_bytes);
  __bf16* WcatT = (__bf16*)(ws + 2 * lt_bytes);
  __bf16* WoT   = (__bf16*)(ws + 2 * lt_bytes + 64 * 256 * 2);

  prep_kernel<<<64, 256, 0, stream>>>(Wl, Wr, Wo, WcatT, WoT);

  ln_proj_kernel<<<384, 256, 128 * K1_STRIDE, stream>>>(
      msa, ln_g, ln_b, WcatT, bl, br, LT, RT);

  dim3 grid2(N_DIM / 8, N_DIM / 8);  // 48 x 48
  size_t smem2 = 2 * 256 * LR_STRIDE + 64 * OT_STRIDE;  // 271360 B < 320 KB
  outer_proj_kernel<<<grid2, 256, smem2, stream>>>(LT, RT, WoT, bo, out);
}
